// TransformerAttention_29703993819437
// MI455X (gfx1250) — matrix operand order repaired and checked by reading, not compiled, not run on hardware
//
#include <hip/hip_runtime.h>
#include <hip/hip_bf16.h>

#define DIM 1024
#define NTOK 4096      // 2 * 2048 tokens
#define SEQ 2048
#define HEADS 16
#define DHEAD 64
#define NQKV 3072
#define NBH 32         // batch * heads

typedef __attribute__((ext_vector_type(16))) __bf16 v16bf;
typedef __attribute__((ext_vector_type(8)))  __bf16 v8bf;
typedef __attribute__((ext_vector_type(4)))  __bf16 v4bf;
typedef __attribute__((ext_vector_type(8)))  float  v8f;
typedef __attribute__((ext_vector_type(4)))  int    v4i;
typedef __attribute__((address_space(1))) v4i gv4i;
typedef __attribute__((address_space(3))) v4i lv4i;

// D = A(16x32 bf16) * B(32x16 bf16) + C(16x16 f32)
static __device__ inline v8f wmma_bf16(v16bf a, v16bf b, v8f c) {
  return __builtin_amdgcn_wmma_f32_16x16x32_bf16(
      /*neg_a=*/false, a, /*neg_b=*/false, b,
      /*c_mod=*/(short)0, c, /*reuse_a=*/false, /*reuse_b=*/false);
}

// A-fragment: lane's halves 0..7 at +0, halves 8..15 at +16 (ISA 16-bit A layout)
static __device__ inline v16bf ldA(const __bf16* lo) {
  union { v16bf v; v8bf h[2]; } u;
  u.h[0] = *(const v8bf*)lo;
  u.h[1] = *(const v8bf*)(lo + 16);
  return u.v;
}
// B-fragment: lane's halves 0..7 at +0, halves 8..15 at +16 (matches A layout; K-major storage)
static __device__ inline v16bf ldB(const __bf16* p) {
  union { v16bf v; v8bf h[2]; } u;
  u.h[0] = *(const v8bf*)p;
  u.h[1] = *(const v8bf*)(p + 16);
  return u.v;
}

// ---- async global->LDS staging (gfx1250 ASYNCcnt path), sync fallback ----
static __device__ inline void copy_b128_to_lds(const __bf16* g, __bf16* l) {
#if __has_builtin(__builtin_amdgcn_global_load_async_to_lds_b128)
  __builtin_amdgcn_global_load_async_to_lds_b128(
      (gv4i*)(v4i*)(void*)g, (lv4i*)(v4i*)(void*)l, 0, 0);
#else
  *(v8bf*)l = *(const v8bf*)g;
#endif
}
static __device__ inline void staging_wait() {
#if __has_builtin(__builtin_amdgcn_global_load_async_to_lds_b128)
#if __has_builtin(__builtin_amdgcn_s_wait_asynccnt)
  __builtin_amdgcn_s_wait_asynccnt(0);
#else
  asm volatile("s_wait_asynccnt 0x0" ::: "memory");
#endif
#endif
}

static __device__ inline float half16_max(float v) {
#pragma unroll
  for (int off = 8; off >= 1; off >>= 1) v = fmaxf(v, __shfl_xor(v, off, 32));
  return v;
}
static __device__ inline float half16_sum(float v) {
#pragma unroll
  for (int off = 8; off >= 1; off >>= 1) v += __shfl_xor(v, off, 32);
  return v;
}

// -------------------- LayerNorm (fp32 in -> bf16 out) --------------------
__global__ __launch_bounds__(256) void ln_kernel(
    const float* __restrict__ x, const float* __restrict__ gamma,
    const float* __restrict__ beta, __bf16* __restrict__ out) {
  const int row = blockIdx.x;
  const int t = threadIdx.x;
  const float4 v = ((const float4*)(x + (size_t)row * DIM))[t];
  float s = v.x + v.y + v.z + v.w;
#pragma unroll
  for (int off = 16; off >= 1; off >>= 1) s += __shfl_xor(s, off, 32);
  __shared__ float red[8], red2[8];
  const int wv = t >> 5, ln = t & 31;
  if (ln == 0) red[wv] = s;
  __syncthreads();
  float tot = 0.f;
#pragma unroll
  for (int i = 0; i < 8; ++i) tot += red[i];
  const float mu = tot * (1.0f / DIM);
  const float d0 = v.x - mu, d1 = v.y - mu, d2 = v.z - mu, d3 = v.w - mu;
  float q = d0 * d0 + d1 * d1 + d2 * d2 + d3 * d3;
#pragma unroll
  for (int off = 16; off >= 1; off >>= 1) q += __shfl_xor(q, off, 32);
  if (ln == 0) red2[wv] = q;
  __syncthreads();
  float vt = 0.f;
#pragma unroll
  for (int i = 0; i < 8; ++i) vt += red2[i];
  const float rstd = rsqrtf(vt * (1.0f / DIM) + 1e-5f);
  const float4 g4 = ((const float4*)gamma)[t];
  const float4 b4 = ((const float4*)beta)[t];
  v4bf o;
  o[0] = (__bf16)(d0 * rstd * g4.x + b4.x);
  o[1] = (__bf16)(d1 * rstd * g4.y + b4.y);
  o[2] = (__bf16)(d2 * rstd * g4.z + b4.z);
  o[3] = (__bf16)(d3 * rstd * g4.w + b4.w);
  *(v4bf*)(out + (size_t)row * DIM + t * 4) = o;
}

// ---------------- fp32 [R,C] -> bf16 transposed [C,R] ----------------
__global__ __launch_bounds__(256) void cvt_transpose_kernel(
    const float* __restrict__ in, __bf16* __restrict__ out, int R, int C) {
  int idx = blockIdx.x * 256 + threadIdx.x;
  if (idx >= R * C) return;
  int r = idx / C, c = idx - r * C;
  out[(size_t)c * R + r] = (__bf16)in[idx];
}

// ---------------- shared 32x64-per-wave GEMM mainloop ----------------
// Block: 8 waves, tile 256 rows x 64 cols. B tile (64x32 K) staged in LDS,
// double-buffered, filled with async global->LDS; A prefetched into registers.
static __device__ inline void gemm32x64_mainloop(
    const __bf16* __restrict__ A, const __bf16* __restrict__ Bt,
    int rowTile, int colTile, __bf16* ldsB /* [2][2048] halves */, v8f* acc) {
  const int tid = threadIdx.x;
  const int lane = tid & 31;
  const int laneLo = lane & 15, laneHi = lane >> 4;
  const __bf16* arow0 = A + (size_t)(rowTile + laneLo) * DIM + laneHi * 8;
  const __bf16* arow1 = arow0 + (size_t)16 * DIM;
  // this thread's 16B slice of the 4KB B tile (exactly 256 slices)
  const int st = tid >> 6;            // fragment 0..3 (16 cols each)
  const int sub = tid & 63;
  const int sLane = sub >> 1, sHalf = sub & 1;
  const __bf16* gsrc = Bt + (size_t)(colTile + 16 * st + (sLane & 15)) * DIM +
                       (sLane >> 4) * 8 + sHalf * 16;
  __bf16* ldst = ldsB + st * 512 + sLane * 16 + sHalf * 8;

  copy_b128_to_lds(gsrc, ldst);            // stage kk = 0 into buffer 0
  v16bf a0 = ldA(arow0);
  v16bf a1 = ldA(arow1);
  for (int kk = 0; kk < DIM; kk += 32) {
    const int buf = (kk >> 5) & 1;
    staging_wait();
    __syncthreads();                       // B tile for kk ready; prev reads done
    v16bf na0 = a0, na1 = a1;
    if (kk + 32 < DIM) {
      copy_b128_to_lds(gsrc + kk + 32, ldst + (buf ^ 1) * 2048);
      na0 = ldA(arow0 + kk + 32);          // register-pipeline A
      na1 = ldA(arow1 + kk + 32);
    }
    const __bf16* lb = ldsB + buf * 2048 + lane * 16;
#pragma unroll
    for (int t = 0; t < 4; ++t) {
      v16bf b = *(const v16bf*)(lb + t * 512);
      acc[t]     = wmma_bf16(a0, b, acc[t]);
      acc[4 + t] = wmma_bf16(a1, b, acc[4 + t]);
    }
    a0 = na0; a1 = na1;
  }
}

// ---------------- QKV GEMM: [NTOK,DIM] x [DIM,NQKV] -> Q/K/Vt ----------------
__global__ __launch_bounds__(256) void qkv_gemm_kernel(
    const __bf16* __restrict__ A,   // xn [NTOK, DIM]
    const __bf16* __restrict__ Bt,  // w_qkv^T [NQKV, DIM]
    __bf16* __restrict__ Qo, __bf16* __restrict__ Ko, __bf16* __restrict__ Vt) {
  __shared__ __align__(32) __bf16 ldsB[2][2048];
  const int lane = threadIdx.x & 31;
  const int wave = threadIdx.x >> 5;
  const int laneLo = lane & 15, laneHi = lane >> 4;
  const int rowTile = blockIdx.y * 256 + wave * 32;
  const int colTile = blockIdx.x * 64;
  v8f acc[8] = {};
  gemm32x64_mainloop(A, Bt, rowTile, colTile, &ldsB[0][0], acc);
  const int which = colTile >> 10;      // 0=Q 1=K 2=V
  const int h = (colTile & 1023) >> 6;  // head (64-col tile stays in one head)
#pragma unroll
  for (int g = 0; g < 2; ++g) {
#pragma unroll
    for (int t = 0; t < 4; ++t) {
#pragma unroll
      for (int v = 0; v < 8; ++v) {
        const int r = rowTile + g * 16 + v + 8 * laneHi;
        const int d = 16 * t + laneLo;
        const int b = r >> 11, nn = r & 2047;
        const int bh = b * HEADS + h;
        const float val = acc[g * 4 + t][v];
        if (which == 0)
          Qo[((size_t)bh * SEQ + nn) * DHEAD + d] = (__bf16)(val * 0.125f);
        else if (which == 1)
          Ko[((size_t)bh * SEQ + nn) * DHEAD + d] = (__bf16)val;
        else
          Vt[((size_t)bh * DHEAD + d) * SEQ + nn] = (__bf16)val;
      }
    }
  }
}

// ---------------- Flash attention per (b,h): softmax(QK^T)V ----------------
__global__ __launch_bounds__(256) void attn_kernel(
    const __bf16* __restrict__ Q, const __bf16* __restrict__ Km,
    const __bf16* __restrict__ Vt, __bf16* __restrict__ AO) {
  __shared__ __align__(32) __bf16 p_lds[8][16 * 32];  // per-wave P tile
  const int lane = threadIdx.x & 31;
  const int wave = threadIdx.x >> 5;
  const int laneLo = lane & 15, laneHi = lane >> 4;
  const int bh = blockIdx.y;
  const int qBase = blockIdx.x * 128 + wave * 16;
  const __bf16* Qp = Q + (size_t)bh * SEQ * DHEAD;
  const __bf16* Kp = Km + (size_t)bh * SEQ * DHEAD;
  const __bf16* Vp = Vt + (size_t)bh * DHEAD * SEQ;
  const __bf16* qrow = Qp + (size_t)(qBase + laneLo) * DHEAD + laneHi * 8;
  const v16bf qa0 = ldA(qrow);
  const v16bf qa1 = ldA(qrow + 32);
  v8f o0 = {}, o1 = {}, o2 = {}, o3 = {};
  float m[8], l[8];
#pragma unroll
  for (int v = 0; v < 8; ++v) { m[v] = -1e30f; l[v] = 0.f; }
  __bf16* pl = &p_lds[wave][0];
  for (int kt = 0; kt < SEQ; kt += 32) {
    const __bf16* kr0 = Kp + (size_t)(kt + laneLo) * DHEAD + laneHi * 8;
    const __bf16* kr1 = kr0 + 16 * DHEAD;
    v8f s0 = {}, s1 = {};
    s0 = wmma_bf16(qa0, ldB(kr0), s0);
    s0 = wmma_bf16(qa1, ldB(kr0 + 32), s0);
    s1 = wmma_bf16(qa0, ldB(kr1), s1);
    s1 = wmma_bf16(qa1, ldB(kr1 + 32), s1);
#pragma unroll
    for (int v = 0; v < 8; ++v) {
      float tm = half16_max(fmaxf(s0[v], s1[v]));
      const float mn = fmaxf(m[v], tm);
      const float alpha = __expf(m[v] - mn);
      m[v] = mn;
      const float p0 = __expf(s0[v] - mn);
      const float p1 = __expf(s1[v] - mn);
      const int rrow = v + 8 * laneHi;
      pl[rrow * 32 + laneLo] = (__bf16)p0;
      pl[rrow * 32 + 16 + laneLo] = (__bf16)p1;
      l[v] = l[v] * alpha + half16_sum(p0 + p1);
      o0[v] *= alpha; o1[v] *= alpha; o2[v] *= alpha; o3[v] *= alpha;
    }
    const v16bf pa = ldA(pl + laneLo * 32 + laneHi * 8);
    const __bf16* vr = Vp + (size_t)laneLo * SEQ + kt + laneHi * 8;
    o0 = wmma_bf16(pa, ldB(vr), o0);
    o1 = wmma_bf16(pa, ldB(vr + (size_t)16 * SEQ), o1);
    o2 = wmma_bf16(pa, ldB(vr + (size_t)32 * SEQ), o2);
    o3 = wmma_bf16(pa, ldB(vr + (size_t)48 * SEQ), o3);
  }
  const int b = bh >> 4, h = bh & 15;
#pragma unroll
  for (int v = 0; v < 8; ++v) {
    const float inv = 1.0f / l[v];
    const size_t rowOut = (size_t)b * SEQ + qBase + v + 8 * laneHi;
    __bf16* op = AO + rowOut * DIM + h * DHEAD + laneLo;
    op[0]  = (__bf16)(o0[v] * inv);
    op[16] = (__bf16)(o1[v] * inv);
    op[32] = (__bf16)(o2[v] * inv);
    op[48] = (__bf16)(o3[v] * inv);
  }
}

// ---------------- Output projection GEMM + bias (fp32 out) ----------------
__global__ __launch_bounds__(256) void out_gemm_kernel(
    const __bf16* __restrict__ A,   // attn out [NTOK, DIM]
    const __bf16* __restrict__ Bt,  // w_out^T [DIM, DIM]
    const float* __restrict__ bias, float* __restrict__ out) {
  __shared__ __align__(32) __bf16 ldsB[2][2048];
  const int lane = threadIdx.x & 31;
  const int wave = threadIdx.x >> 5;
  const int laneLo = lane & 15, laneHi = lane >> 4;
  const int rowTile = blockIdx.y * 256 + wave * 32;
  const int colTile = blockIdx.x * 64;
  v8f acc[8] = {};
  gemm32x64_mainloop(A, Bt, rowTile, colTile, &ldsB[0][0], acc);
#pragma unroll
  for (int g = 0; g < 2; ++g) {
#pragma unroll
    for (int t = 0; t < 4; ++t) {
#pragma unroll
      for (int v = 0; v < 8; ++v) {
        const int r = rowTile + g * 16 + v + 8 * laneHi;
        const int c = colTile + 16 * t + laneLo;
        out[(size_t)r * DIM + c] = acc[g * 4 + t][v] + bias[c];
      }
    }
  }
}

extern "C" void kernel_launch(void* const* d_in, const int* in_sizes, int n_in,
                              void* d_out, int out_size, void* d_ws, size_t ws_size,
                              hipStream_t stream) {
  (void)in_sizes; (void)n_in; (void)out_size; (void)ws_size;
  const float* x     = (const float*)d_in[0];
  const float* gamma = (const float*)d_in[1];
  const float* beta  = (const float*)d_in[2];
  const float* wqkv  = (const float*)d_in[3];
  const float* wout  = (const float*)d_in[4];
  const float* bout  = (const float*)d_in[5];
  float* out = (float*)d_out;

  char* ws = (char*)d_ws;
  size_t off = 0;
  auto take = [&](size_t elems) {
    __bf16* p = (__bf16*)(ws + off);
    off += elems * sizeof(__bf16);
    return p;
  };
  __bf16* xn    = take((size_t)NTOK * DIM);          // 8 MB
  __bf16* wqkvt = take((size_t)NQKV * DIM);          // 6 MB
  __bf16* woutt = take((size_t)DIM * DIM);           // 2 MB
  __bf16* Qb    = take((size_t)NBH * SEQ * DHEAD);   // 8 MB
  __bf16* Kb    = take((size_t)NBH * SEQ * DHEAD);   // 8 MB
  __bf16* Vtb   = take((size_t)NBH * SEQ * DHEAD);   // 8 MB
  __bf16* AO    = take((size_t)NTOK * DIM);          // 8 MB

  ln_kernel<<<NTOK, 256, 0, stream>>>(x, gamma, beta, xn);
  cvt_transpose_kernel<<<(DIM * NQKV + 255) / 256, 256, 0, stream>>>(wqkv, wqkvt, DIM, NQKV);
  cvt_transpose_kernel<<<(DIM * DIM + 255) / 256, 256, 0, stream>>>(wout, woutt, DIM, DIM);
  qkv_gemm_kernel<<<dim3(NQKV / 64, NTOK / 256), 256, 0, stream>>>(xn, wqkvt, Qb, Kb, Vtb);
  attn_kernel<<<dim3(SEQ / 128, NBH), 256, 0, stream>>>(Qb, Kb, Vtb, AO);
  out_gemm_kernel<<<dim3(DIM / 64, NTOK / 256), 256, 0, stream>>>(AO, woutt, bout, out);
}
